// NCNPredictor_446676599133
// MI455X (gfx1250) — matrix operand order repaired and checked by reading, not compiled, not run on hardware
//
#include <hip/hip_runtime.h>

// NCN link predictor for gfx1250 (MI455X).
// Dominant cost: xcn = (adj[src] AND adj[dst]) @ x2  -> 34.4 GFLOP, 512 MB adj stream.
// HBM-bound (~22us @ 23.3 TB/s) once the GEMM runs on v_wmma_f32_16x16x32_bf16 and
// x2 (4 MB bf16, packed into WMMA B-fragment layout) stays L2-resident.
// adj rows are staged global->LDS with GLOBAL_LOAD_ASYNC_TO_LDS_B128 (ASYNCcnt
// pipeline) when the toolchain exposes the builtin; otherwise NT loads.

#define N_NODES 8192
#define H_DIM   256
#define E_EDGES 8192

typedef __attribute__((ext_vector_type(16))) __bf16 v16bf;
typedef __attribute__((ext_vector_type(8)))  __bf16 v8bf;
typedef __attribute__((ext_vector_type(2)))  __bf16 v2bf;
typedef __attribute__((ext_vector_type(8)))  float  v8f;
typedef __attribute__((ext_vector_type(4)))  float  f4;
typedef __attribute__((ext_vector_type(4)))  int    v4i;

#if defined(__has_builtin)
#if __has_builtin(__builtin_amdgcn_global_load_async_to_lds_b128) && \
    __has_builtin(__builtin_amdgcn_s_wait_asynccnt)
#define K3_ASYNC 1
#endif
#endif

// Builtin signature (from clang diagnostics): pointee type is int4, global arg in
// addrspace(1), LDS arg in addrspace(3); 4 args: (gaddr, ldsaddr, offset, cpol).
#define AS1G(p) ((__attribute__((address_space(1))) v4i*)(p))
#define AS3L(p) ((__attribute__((address_space(3))) v4i*)(p))

// Native bf16 conversion: let the backend pick v_cvt(_pk)_bf16_f32.
__device__ __forceinline__ __bf16 f2bf(float f) { return (__bf16)f; }

__device__ __forceinline__ v2bf f2bf_pk(float lo, float hi) {
#if defined(__has_builtin)
#if __has_builtin(__builtin_amdgcn_cvt_pk_bf16_f32)
  return __builtin_amdgcn_cvt_pk_bf16_f32(lo, hi);
#else
  v2bf r; r[0] = (__bf16)lo; r[1] = (__bf16)hi; return r;
#endif
#else
  v2bf r; r[0] = (__bf16)lo; r[1] = (__bf16)hi; return r;
#endif
}

__device__ __forceinline__ v8f wmma_bf16(v16bf a, v16bf b, v8f c) {
  // 8 args: (neg_a, A, neg_b, B, c_mod, C, reuse_a, reuse_b)
  return __builtin_amdgcn_wmma_f32_16x16x32_bf16(false, a, false, b, (short)0, c,
                                                 false, false);
}

union V16u { v16bf v; v2bf p[8]; };
union V8u  { v8bf  v; v2bf p[4]; };

// A-fragment per-lane build from fp32: p0 -> elements 0..7 (k-run), p1 -> 8..15.
__device__ __forceinline__ v16bf packA_f32(const float* __restrict__ p0,
                                           const float* __restrict__ p1) {
  f4 a = *(const f4*)p0;
  f4 b = *(const f4*)(p0 + 4);
  f4 c = *(const f4*)p1;
  f4 d = *(const f4*)(p1 + 4);
  V16u r;
  r.p[0] = f2bf_pk(a.x, a.y); r.p[1] = f2bf_pk(a.z, a.w);
  r.p[2] = f2bf_pk(b.x, b.y); r.p[3] = f2bf_pk(b.z, b.w);
  r.p[4] = f2bf_pk(c.x, c.y); r.p[5] = f2bf_pk(c.z, c.w);
  r.p[6] = f2bf_pk(d.x, d.y); r.p[7] = f2bf_pk(d.z, d.w);
  return r.v;
}

// ---------------------------------------------------------------------------
// K1: pack a 256x256 fp32 weight (K x N, row-major) into bf16 B-fragment blocks.
// Block (kt,nt) is 32x16; element index = ((kt*16+nt)*32 + lane)*16 + e,
// with lane = half*16 + n, k = kt*32 + (e&7) + half*8 + (e>>3)*16.
// ---------------------------------------------------------------------------
__global__ void pack_weight_kernel(const float* __restrict__ w,
                                   __bf16* __restrict__ dst) {
  int t    = blockIdx.x * blockDim.x + threadIdx.x;  // 0..65535
  int e    = t & 15;
  int lane = (t >> 4) & 31;
  int blk  = t >> 9;                                  // kt*16 + nt
  int nt = blk & 15, kt = blk >> 4;
  int half = lane >> 4, n = lane & 15;
  int k = kt * 32 + (e & 7) + half * 8 + (e >> 3) * 16;
  dst[t] = f2bf(w[k * H_DIM + nt * 16 + n]);
}

// ---------------------------------------------------------------------------
// K2: x2 = x + x@xlin_w + b, written as bf16 in packed B-fragment layout
// (K = node dim 8192, N = H 256) for the cn GEMM.
// ---------------------------------------------------------------------------
__global__ void __launch_bounds__(256)
x2_gemm_kernel(const float* __restrict__ x,
               const __bf16* __restrict__ wpack,
               const float* __restrict__ bias,
               __bf16* __restrict__ x2p) {
  const int lane = threadIdx.x & 31, wave = threadIdx.x >> 5;
  const int col = lane & 15, half = lane >> 4;
  const int mBase = blockIdx.x * 16;
  const int nt0 = wave * 2;                    // 8 waves x 2 n-tiles = 16 tiles
  const float* rowp = x + (size_t)(mBase + col) * H_DIM;   // A row for this lane
  v8f acc[2] = {};
#pragma unroll
  for (int ks = 0; ks < 8; ++ks) {
    const int k0 = ks * 32 + half * 8;
    v16bf a = packA_f32(rowp + k0, rowp + k0 + 16);
#pragma unroll
    for (int j = 0; j < 2; ++j) {
      const int nt = nt0 + j;
      v16bf b = *(const v16bf*)(wpack + (((size_t)ks * 16 + nt) * 32 + lane) * 16);
      acc[j] = wmma_bf16(a, b, acc[j]);
    }
  }
#pragma unroll
  for (int j = 0; j < 2; ++j) {
    const int nt = nt0 + j;
    const int n  = nt * 16 + col;
    const float bb = bias[n];
#pragma unroll
    for (int r = 0; r < 8; ++r) {
      const int mg = mBase + r + half * 8;               // global node row
      float v = acc[j][r] + bb + x[(size_t)mg * H_DIM + n];
      const int kt = mg >> 5, kq = mg & 31;
      const int hB = (kq >> 3) & 1, eB = (kq & 7) | ((kq >> 4) << 3);  // B layout: k = (e&7) + (e>=8)*16 + half*8
      x2p[(((size_t)kt * 16 + nt) * 32 + hB * 16 + col) * 16 + eB] = f2bf(v);
    }
  }
}

// ---------------------------------------------------------------------------
// K3 (hot): xcn[e,:] = (adj[src[e]] * adj[dst[e]]) @ x2.
// 32 edges / workgroup, 8 waves, each wave: 2 M-tiles x 2 N-tiles accumulators.
// K-chunks of 128. Async path: adj rows DMAed global->LDS (ASYNCcnt), next
// chunk's fill overlaps current chunk's WMMA. Fallback: NT loads + prefetch.
// ---------------------------------------------------------------------------
__global__ void __launch_bounds__(256)
cn_gemm_kernel(const float* __restrict__ adj,
               const int* __restrict__ tar,
               const __bf16* __restrict__ x2p,
               __bf16* __restrict__ xcnp) {
  __shared__ __align__(32) __bf16 ldsA[2][4][32][16];   // [mtile][kstep][lane][e]
#if defined(K3_ASYNC)
  __shared__ __align__(16) float rawS[32 * 128];        // staged adj[src] chunk
  __shared__ __align__(16) float rawD[32 * 128];        // staged adj[dst] chunk
#endif
  const int t = threadIdx.x;
  const int lane = t & 31, wave = t >> 5;
  const int col = lane & 15, half = lane >> 4;
  const int eBase = blockIdx.x * 32;
  const int nt0 = wave * 2;
  // cn producer mapping: thread -> one edge row, 16 consecutive k per chunk
  const int mRow = t >> 3;            // 0..31
  const int kOff = (t & 7) * 16;      // 0..112
  const int mt = mRow >> 4, mlow = mRow & 15;
  const int srcIdx = tar[eBase + mRow];
  const int dstIdx = tar[E_EDGES + eBase + mRow];
  const float* srcRow = adj + (size_t)srcIdx * N_NODES;
  const float* dstRow = adj + (size_t)dstIdx * N_NODES;
  v8f acc[2][2] = {};

#if defined(K3_ASYNC)
  {  // kick off chunk 0 fill
    float* ls = &rawS[mRow * 128 + kOff];
    float* ld2 = &rawD[mRow * 128 + kOff];
#pragma unroll
    for (int i = 0; i < 4; ++i) {
      __builtin_amdgcn_global_load_async_to_lds_b128(AS1G(srcRow + kOff + 4 * i),
                                                     AS3L(ls + 4 * i), 0, 0);
      __builtin_amdgcn_global_load_async_to_lds_b128(AS1G(dstRow + kOff + 4 * i),
                                                     AS3L(ld2 + 4 * i), 0, 0);
    }
  }
#endif

  for (int c = 0; c < 64; ++c) {
    const int kBase = c << 7;
    f4 s[4], d[4];
#if defined(K3_ASYNC)
    __builtin_amdgcn_s_wait_asynccnt(0);   // this wave's fill done
    __syncthreads();                       // all waves' fills done; prev WMMA done
    {
      const f4* ps = (const f4*)&rawS[mRow * 128 + kOff];
      const f4* pd = (const f4*)&rawD[mRow * 128 + kOff];
#pragma unroll
      for (int i = 0; i < 4; ++i) { s[i] = ps[i]; d[i] = pd[i]; }
    }
    __syncthreads();                       // raw buffer fully consumed
    if (c < 63) {                          // overlap next fill with WMMA below
      float* ls = &rawS[mRow * 128 + kOff];
      float* ld2 = &rawD[mRow * 128 + kOff];
      const float* gs = srcRow + kBase + 128 + kOff;
      const float* gd = dstRow + kBase + 128 + kOff;
#pragma unroll
      for (int i = 0; i < 4; ++i) {
        __builtin_amdgcn_global_load_async_to_lds_b128(AS1G(gs + 4 * i),
                                                       AS3L(ls + 4 * i), 0, 0);
        __builtin_amdgcn_global_load_async_to_lds_b128(AS1G(gd + 4 * i),
                                                       AS3L(ld2 + 4 * i), 0, 0);
      }
    }
#else
    __syncthreads();                       // protect LDS frag reads of prev chunk
#pragma unroll
    for (int i = 0; i < 4; ++i) {
      s[i] = __builtin_nontemporal_load((const f4*)(srcRow + kBase + kOff) + i);
      d[i] = __builtin_nontemporal_load((const f4*)(dstRow + kBase + kOff) + i);
    }
#endif
    // build cn bf16 A-fragments in LDS
#pragma unroll
    for (int g = 0; g < 2; ++g) {
      const int kk = kOff + g * 8;
      const f4 s0 = s[2 * g], s1 = s[2 * g + 1];
      const f4 d0 = d[2 * g], d1 = d[2 * g + 1];
      const int kstep = kk >> 5, kq = kk & 31;
      const int hA = (kq >> 3) & 1;        // A layout: k = (e&7) + (e>=8)*16 + half*8
      const int e0 = (kq >> 4) << 3;
      V8u cv;
      cv.p[0] = f2bf_pk(s0.x * d0.x, s0.y * d0.y);
      cv.p[1] = f2bf_pk(s0.z * d0.z, s0.w * d0.w);
      cv.p[2] = f2bf_pk(s1.x * d1.x, s1.y * d1.y);
      cv.p[3] = f2bf_pk(s1.z * d1.z, s1.w * d1.w);
      *(v8bf*)&ldsA[mt][kstep][hA * 16 + mlow][e0] = cv.v;
    }
    __syncthreads();
#if !defined(K3_ASYNC)
    if (c < 63) {
      __builtin_prefetch(srcRow + kBase + 128 + kOff, 0, 1);
      __builtin_prefetch(dstRow + kBase + 128 + kOff, 0, 1);
    }
#endif
#pragma unroll
    for (int ks = 0; ks < 4; ++ks) {
      v16bf a0 = *(const v16bf*)&ldsA[0][ks][lane][0];
      v16bf a1 = *(const v16bf*)&ldsA[1][ks][lane][0];
      const int ktg = (kBase >> 5) + ks;
#pragma unroll
      for (int j = 0; j < 2; ++j) {
        const int nt = nt0 + j;
        v16bf b = *(const v16bf*)(x2p + (((size_t)ktg * 16 + nt) * 32 + lane) * 16);
        acc[0][j] = wmma_bf16(a0, b, acc[0][j]);
        acc[1][j] = wmma_bf16(a1, b, acc[1][j]);
      }
    }
  }
  // store xcn as bf16 row-major [E, H] (consumed as A-fragments by K4)
#pragma unroll
  for (int m2 = 0; m2 < 2; ++m2)
#pragma unroll
    for (int j = 0; j < 2; ++j) {
      const int nt = nt0 + j;
#pragma unroll
      for (int r = 0; r < 8; ++r) {
        const int edge = eBase + m2 * 16 + r + half * 8;
        xcnp[(size_t)edge * H_DIM + nt * 16 + col] = f2bf(acc[m2][j][r]);
      }
    }
}

// ---------------------------------------------------------------------------
// K4: per 16-edge tile: xij0 = relu(xi@Wi + bi + xj@Wj + bj)  (WMMA, gathers)
//     xij  = xij0@Wf + bf ; xcnl = xcn@Wcn + bcn  (WMMA)
//     xs = xcnl*beta + xij ; out = -log_sigmoid(sign * (xs@wS + bS))
// ---------------------------------------------------------------------------
__global__ void __launch_bounds__(256)
edge_epilogue_kernel(const float* __restrict__ x,
                     const int* __restrict__ tar,
                     const __bf16* __restrict__ xcnp,
                     const __bf16* __restrict__ wpI,
                     const __bf16* __restrict__ wpJ,
                     const __bf16* __restrict__ wpF,
                     const __bf16* __restrict__ wpC,
                     const float* __restrict__ biasI,
                     const float* __restrict__ biasJ,
                     const float* __restrict__ biasF,
                     const float* __restrict__ biasC,
                     const float* __restrict__ wS,
                     const float* __restrict__ bS,
                     const float* __restrict__ beta,
                     const int* __restrict__ boolen,
                     float* __restrict__ out) {
  __shared__ __align__(32) __bf16 ldsA[8][32][16];   // xij0 as A-fragments
  __shared__ float sdot[16];
  const int lane = threadIdx.x & 31, wave = threadIdx.x >> 5;
  const int col = lane & 15, half = lane >> 4;
  const int eBase = blockIdx.x * 16;
  const int nt0 = wave * 2;
  const int mlow = lane & 15;
  const int srcIdx = tar[eBase + mlow];
  const int dstIdx = tar[E_EDGES + eBase + mlow];
  const float* rs = x + (size_t)srcIdx * H_DIM;
  const float* rd = x + (size_t)dstIdx * H_DIM;
  if (threadIdx.x < 16) sdot[threadIdx.x] = 0.0f;

  // phase 1
  v8f accP[2] = {};
#pragma unroll
  for (int ks = 0; ks < 8; ++ks) {
    const int k0 = ks * 32 + half * 8;
    v16bf aS = packA_f32(rs + k0, rs + k0 + 16);
    v16bf aD = packA_f32(rd + k0, rd + k0 + 16);
#pragma unroll
    for (int j = 0; j < 2; ++j) {
      const int nt = nt0 + j;
      const size_t boff = (((size_t)ks * 16 + nt) * 32 + lane) * 16;
      accP[j] = wmma_bf16(aS, *(const v16bf*)(wpI + boff), accP[j]);
      accP[j] = wmma_bf16(aD, *(const v16bf*)(wpJ + boff), accP[j]);
    }
  }
#pragma unroll
  for (int j = 0; j < 2; ++j) {
    const int nt = nt0 + j;
    const int n  = nt * 16 + col;
    const float bb = biasI[n] + biasJ[n];
    const int ks2 = n >> 5, kq = n & 31;       // n becomes K index of next GEMM
    const int hA = (kq >> 3) & 1;
    const int eA = (kq & 7) | ((kq >> 4) << 3);
#pragma unroll
    for (int r = 0; r < 8; ++r) {
      const int mrow = r + half * 8;
      float v = fmaxf(accP[j][r] + bb, 0.0f);
      ldsA[ks2][hA * 16 + mrow][eA] = f2bf(v);
    }
  }
  __syncthreads();

  // phase 2
  v8f acc2[2] = {};   // xij0 @ Wfinal
  v8f acc3[2] = {};   // xcn  @ Wcn
  const __bf16* rc = xcnp + (size_t)(eBase + mlow) * H_DIM;
#pragma unroll
  for (int ks = 0; ks < 8; ++ks) {
    const int k0 = ks * 32 + half * 8;
    v16bf a2 = *(const v16bf*)&ldsA[ks][lane][0];
    v8bf lo = *(const v8bf*)(rc + k0);
    v8bf hi = *(const v8bf*)(rc + k0 + 16);
    v16bf a3;
#pragma unroll
    for (int i = 0; i < 8; ++i) { a3[i] = lo[i]; a3[8 + i] = hi[i]; }
#pragma unroll
    for (int j = 0; j < 2; ++j) {
      const int nt = nt0 + j;
      const size_t boff = (((size_t)ks * 16 + nt) * 32 + lane) * 16;
      acc2[j] = wmma_bf16(a2, *(const v16bf*)(wpF + boff), acc2[j]);
      acc3[j] = wmma_bf16(a3, *(const v16bf*)(wpC + boff), acc3[j]);
    }
  }
  const float betaV = beta[0];
#pragma unroll
  for (int j = 0; j < 2; ++j) {
    const int nt = nt0 + j;
    const int n  = nt * 16 + col;
    const float bc = biasC[n], bf = biasF[n], wn = wS[n];
#pragma unroll
    for (int r = 0; r < 8; ++r) {
      float xs = (acc3[j][r] + bc) * betaV + (acc2[j][r] + bf);
      atomicAdd(&sdot[r + half * 8], xs * wn);   // ds_add_f32
    }
  }
  __syncthreads();
  if (threadIdx.x < 16) {
    float z  = sdot[threadIdx.x] + bS[0];
    float zz = (boolen[0] != 0) ? z : -z;
    // -log_sigmoid(zz) = softplus(-zz), numerically stable
    out[eBase + threadIdx.x] = fmaxf(-zz, 0.0f) + log1pf(expf(-fabsf(zz)));
  }
}

// ---------------------------------------------------------------------------
extern "C" void kernel_launch(void* const* d_in, const int* in_sizes, int n_in,
                              void* d_out, int out_size, void* d_ws, size_t ws_size,
                              hipStream_t stream) {
  (void)in_sizes; (void)n_in; (void)out_size; (void)ws_size;
  const float* x         = (const float*)d_in[0];
  const float* adj       = (const float*)d_in[1];
  const int*   tar       = (const int*)d_in[2];
  const float* xlin_w    = (const float*)d_in[3];
  const float* xlin_b    = (const float*)d_in[4];
  const float* xcnlin_w  = (const float*)d_in[5];
  const float* xcnlin_b  = (const float*)d_in[6];
  const float* xijlini_w = (const float*)d_in[7];
  const float* xijlini_b = (const float*)d_in[8];
  const float* xijlinj_w = (const float*)d_in[9];
  const float* xijlinj_b = (const float*)d_in[10];
  const float* xijfinal_w= (const float*)d_in[11];
  const float* xijfinal_b= (const float*)d_in[12];
  const float* xslin_w   = (const float*)d_in[13];
  const float* xslin_b   = (const float*)d_in[14];
  const float* beta      = (const float*)d_in[15];
  const int*   boolen    = (const int*)d_in[16];

  char* ws = (char*)d_ws;
  __bf16* wp   = (__bf16*)ws;                                       // 5 x 64K elems
  __bf16* x2p  = (__bf16*)(ws + (size_t)5 * 65536 * 2);             // 4 MB
  __bf16* xcnp = (__bf16*)(ws + ((size_t)5 * 65536 + (size_t)N_NODES * H_DIM) * 2);

  pack_weight_kernel<<<256, 256, 0, stream>>>(xlin_w,     wp + 0 * 65536);
  pack_weight_kernel<<<256, 256, 0, stream>>>(xcnlin_w,   wp + 1 * 65536);
  pack_weight_kernel<<<256, 256, 0, stream>>>(xijlini_w,  wp + 2 * 65536);
  pack_weight_kernel<<<256, 256, 0, stream>>>(xijlinj_w,  wp + 3 * 65536);
  pack_weight_kernel<<<256, 256, 0, stream>>>(xijfinal_w, wp + 4 * 65536);

  x2_gemm_kernel<<<N_NODES / 16, 256, 0, stream>>>(x, wp, xlin_b, x2p);
  cn_gemm_kernel<<<E_EDGES / 32, 256, 0, stream>>>(adj, tar, x2p, xcnp);
  edge_epilogue_kernel<<<E_EDGES / 16, 256, 0, stream>>>(
      x, tar, xcnp,
      wp + 2 * 65536, wp + 3 * 65536, wp + 4 * 65536, wp + 1 * 65536,
      xijlini_b, xijlinj_b, xijfinal_b, xcnlin_b,
      xslin_w, xslin_b, beta, boolen, (float*)d_out);
}
